// Double_SSM_Block_Custom_Channel_59382217834751
// MI455X (gfx1250) — matrix operand order repaired and checked by reading, not compiled, not run on hardware
//
#include <hip/hip_runtime.h>
#include <hip/hip_bf16.h>
#include <math.h>

typedef __attribute__((ext_vector_type(16))) _Float16 v16h;
typedef __attribute__((ext_vector_type(8)))  _Float16 v8h;
typedef __attribute__((ext_vector_type(8)))  float    v8f;

#define BATCH  4
#define SEQ    16384
#define MTOT   (BATCH*SEQ)   // 65536 tokens
#define DM     64
#define DI2    256           // 2*DI
#define DI     128
#define DSTATE 16
#define DBL_LD 48            // 36 (dt_rank + 2*16) padded to 48 for 16-wide tiles

typedef __attribute__((address_space(1))) int as1_int;
typedef __attribute__((address_space(3))) int as3_int;

__device__ __forceinline__ float silu_f(float x) {
  return x / (1.f + __expf(-x));
}

// ---- CDNA5 async global->LDS copy (ASYNCcnt-tracked) ----
__device__ __forceinline__ void async_copy_b32(float* lds_dst, const float* gsrc) {
#if __has_builtin(__builtin_amdgcn_global_load_async_to_lds_b32)
  __builtin_amdgcn_global_load_async_to_lds_b32(
      (as1_int*)gsrc, (as3_int*)lds_dst, 0, 0);
#else
  unsigned loff = (unsigned)(size_t)(__attribute__((address_space(3))) char*)lds_dst;
  asm volatile("global_load_async_to_lds_b32 %0, %1, off"
               :: "v"(loff), "v"((unsigned long long)(size_t)gsrc) : "memory");
#endif
}

__device__ __forceinline__ void wait_asynccnt0() {
#if __has_builtin(__builtin_amdgcn_s_wait_asynccnt)
  __builtin_amdgcn_s_wait_asynccnt(0);
#else
  asm volatile("s_wait_asynccnt 0x0" ::: "memory");
#endif
}

// ---------- fp32 -> f16 bulk convert (weights) ----------
__global__ void k_cvt_f16(const float* __restrict__ in, _Float16* __restrict__ out, int n) {
  int i = blockIdx.x * blockDim.x + threadIdx.x;
  if (i < n) out[i] = (_Float16)in[i];
}

// ---------- (b,c,h,w) -> Xh[m = b*L + h*128 + w, c] as f16 (GEMM A operand) ----------
__global__ void k_transpose_in(const float* __restrict__ x, _Float16* __restrict__ Xh) {
  int idx = blockIdx.x * blockDim.x + threadIdx.x;   // over MTOT*DM
  if (idx >= MTOT * DM) return;
  int c = idx & 63;
  int m = idx >> 6;
  int b = m >> 14;
  int l = m & 16383;
  int h = l >> 7, w = l & 127;
  Xh[idx] = (_Float16)x[(((b * 64 + c) * 128) + h) * 128 + w];
}

// ---------- generic WMMA GEMM: C[M,N] = A[M,K] * W^T, all-f16 operands ----------
// One wave32 per 16x16 tile; K consumed 32/chunk via v_wmma_f32_16x16x32_f16.
template<int K>
__global__ void k_gemm_wmma(const _Float16* __restrict__ A, int lda,
                            const _Float16* __restrict__ W, int Nvalid,
                            float* __restrict__ C, int ldc,
                            int ntilesN, int ntiles) {
  int tile = blockIdx.x * (blockDim.x >> 5) + (threadIdx.x >> 5);
  if (tile >= ntiles) return;                 // whole-wave exit keeps EXEC all-1 for WMMA
  int lane = threadIdx.x & 31;
  int m0 = (tile / ntilesN) * 16;
  int n0 = (tile % ntilesN) * 16;
  int lg = lane >> 4;                         // lane group 0/1
  int mrow = m0 + (lane & 15);
  int ncol = n0 + (lane & 15);
  v8f acc = {};
  #pragma unroll
  for (int kk = 0; kk < K; kk += 32) {
    // A frag 16x32: lane l<16 holds K {0..7,16..23}; l>=16 holds {8..15,24..31}
    const _Float16* Arow = A + (size_t)mrow * lda;
    int kbaseA = kk + lg * 8;
    v8h alo = *(const v8h*)(Arow + kbaseA);        // 16B -> global_load_b128
    v8h ahi = *(const v8h*)(Arow + kbaseA + 16);
    v16h a;
    #pragma unroll
    for (int i = 0; i < 8; ++i) { a[i] = alo[i]; a[8 + i] = ahi[i]; }
    // B frag 32x16: lane = column N, same K mapping as A ({0..7,16..23} / {8..15,24..31})
    v16h b;
    int kbaseB = kk + lg * 8;
    if (ncol < Nvalid) {
      const _Float16* Wrow = W + (size_t)ncol * K;  // B[k,n] = W[n*K + k]
      v8h b0 = *(const v8h*)(Wrow + kbaseB);
      v8h b1 = *(const v8h*)(Wrow + kbaseB + 16);
      #pragma unroll
      for (int i = 0; i < 8; ++i) { b[i] = b0[i]; b[8 + i] = b1[i]; }
    } else {
      #pragma unroll
      for (int i = 0; i < 16; ++i) b[i] = (_Float16)0.f;
    }
    acc = __builtin_amdgcn_wmma_f32_16x16x32_f16(false, a, false, b,
                                                 (short)0, acc, false, false);
  }
  int rbase = m0 + lg * 8;                    // C/D: VGPR r -> row +r (+8 for lanes 16..31)
  #pragma unroll
  for (int r = 0; r < 8; ++r)
    C[(size_t)(rbase + r) * ldc + ncol] = acc[r];
}

// ---------- causal depthwise conv (4 taps) + SiLU; emits f32 (scan) + f16 (GEMM) ----------
__global__ void k_conv_silu(const float* __restrict__ XZ, const float* __restrict__ cw,
                            const float* __restrict__ cb,
                            float* __restrict__ XC, _Float16* __restrict__ XCh) {
  int idx = blockIdx.x * blockDim.x + threadIdx.x;   // over MTOT*DI
  if (idx >= MTOT * DI) return;
  int d = idx & 127;
  int m = idx >> 7;
  int l = m & 16383;
  float s = cb[d];
  #pragma unroll
  for (int k = 0; k < 4; ++k) {
    int lt = l + k - 3;                      // causal pad of 3
    if (lt >= 0) s += cw[d * 4 + k] * XZ[(size_t)(m + k - 3) * DI2 + d];
  }
  float v = silu_f(s);
  XC[idx]  = v;
  XCh[idx] = (_Float16)v;
}

// ---------- selective scan with async double-buffered dbl rows ----------
__global__ void k_scan(const float* __restrict__ DBLp, const float* __restrict__ XC,
                       const float* __restrict__ XZ,
                       const float* __restrict__ dt_w, const float* __restrict__ dt_b,
                       const float* __restrict__ A_log, const float* __restrict__ Dp,
                       _Float16* __restrict__ YGh) {
  __shared__ float sRow[2][DBL_LD];
  int b = blockIdx.x;          // one block per batch
  int d = threadIdx.x;         // 128 channels
  float Acol[DSTATE];
  #pragma unroll
  for (int n = 0; n < DSTATE; ++n) Acol[n] = -__expf(A_log[d * DSTATE + n]);
  float w0 = dt_w[d * 4 + 0], w1 = dt_w[d * 4 + 1];
  float w2 = dt_w[d * 4 + 2], w3 = dt_w[d * 4 + 3];
  float bias = dt_b[d], Dd = Dp[d];
  float h[DSTATE];
  #pragma unroll
  for (int n = 0; n < DSTATE; ++n) h[n] = 0.f;

  // prologue: row 0 synchronously
  if (d < DBL_LD) sRow[0][d] = DBLp[(size_t)b * SEQ * DBL_LD + d];
  __syncthreads();

  for (int l = 0; l < SEQ; ++l) {
    int m = b * SEQ + l;
    int cur = l & 1, nxt = cur ^ 1;
    // async-stage next dbl row into the other LDS buffer (overlaps with compute)
    if (l + 1 < SEQ) {
      if (d < DBL_LD)
        async_copy_b32(&sRow[nxt][d], DBLp + (size_t)(m + 1) * DBL_LD + d);
      __builtin_prefetch(&XC[(size_t)(m + 1) * DI + d], 0, 0);        // next xc
      __builtin_prefetch(&XZ[(size_t)(m + 1) * DI2 + DI + d], 0, 0);  // next z
    }
    const float* row = sRow[cur];
    float v  = row[0] * w0 + row[1] * w1 + row[2] * w2 + row[3] * w3 + bias;
    float dt = (v > 20.f) ? v : log1pf(__expf(v));      // softplus
    float xv = XC[(size_t)m * DI + d];
    float du = dt * xv;
    float y = 0.f;
    #pragma unroll
    for (int n = 0; n < DSTATE; ++n) {
      h[n] = __expf(dt * Acol[n]) * h[n] + du * row[4 + n];
      y += h[n] * row[20 + n];
    }
    y += xv * Dd;
    float z = XZ[(size_t)m * DI2 + DI + d];
    y *= silu_f(z);
    YGh[(size_t)m * DI + d] = (_Float16)y;
    wait_asynccnt0();          // next row landed in LDS
    __syncthreads();
  }
}

// ---------- LayerNorm(64)+ReLU; wave32 per row; f16 out (next GEMM) or f32 scatter ----------
__global__ void k_ln_relu(const float* __restrict__ IN, const float* __restrict__ w,
                          const float* __restrict__ bb,
                          _Float16* __restrict__ OUTH, float* __restrict__ OUTF,
                          int scatter) {
  int wave = threadIdx.x >> 5;
  int lane = threadIdx.x & 31;
  int m = blockIdx.x * (blockDim.x >> 5) + wave;
  if (m >= MTOT) return;
  const float* row = IN + (size_t)m * DM;
  float v0 = row[lane], v1 = row[lane + 32];
  float s = v0 + v1, ss = v0 * v0 + v1 * v1;
  #pragma unroll
  for (int off = 16; off >= 1; off >>= 1) {
    s  += __shfl_xor(s, off, 32);
    ss += __shfl_xor(ss, off, 32);
  }
  float mu  = s * (1.f / 64.f);
  float var = ss * (1.f / 64.f) - mu * mu;
  float rs  = rsqrtf(var + 1e-5f);
  float o0 = fmaxf((v0 - mu) * rs * w[lane]      + bb[lane],      0.f);
  float o1 = fmaxf((v1 - mu) * rs * w[lane + 32] + bb[lane + 32], 0.f);
  if (!scatter) {
    OUTH[(size_t)m * DM + lane]      = (_Float16)o0;
    OUTH[(size_t)m * DM + lane + 32] = (_Float16)o1;
  } else {
    // final layout: out[b, c, w, h] = t[b, l=h*128+w, c]
    int b = m >> 14; int l = m & 16383; int hh = l >> 7; int ww = l & 127;
    OUTF[(((size_t)b * 64 + lane)      * 128 + ww) * 128 + hh] = o0;
    OUTF[(((size_t)b * 64 + lane + 32) * 128 + ww) * 128 + hh] = o1;
  }
}

extern "C" void kernel_launch(void* const* d_in, const int* in_sizes, int n_in,
                              void* d_out, int out_size, void* d_ws, size_t ws_size,
                              hipStream_t stream) {
  const float* p[23];
  for (int i = 0; i < 23 && i < n_in; ++i) p[i] = (const float*)d_in[i];
  const float* x = p[0];

  // workspace layout (~172 MB; fits MI455X's 192 MB L2)
  char* ws = (char*)d_ws;
  size_t off = 0;
  float*    XZ   = (float*)(ws + off);    off += (size_t)MTOT * DI2    * 4;
  float*    XC   = (float*)(ws + off);    off += (size_t)MTOT * DI     * 4;
  float*    DBLb = (float*)(ws + off);    off += (size_t)MTOT * DBL_LD * 4;
  float*    O1   = (float*)(ws + off);    off += (size_t)MTOT * DM     * 4;
  _Float16* Xh   = (_Float16*)(ws + off); off += (size_t)MTOT * DM     * 2;
  _Float16* XCh  = (_Float16*)(ws + off); off += (size_t)MTOT * DI     * 2;
  _Float16* YGh  = (_Float16*)(ws + off); off += (size_t)MTOT * DI     * 2;
  _Float16* Wih  = (_Float16*)(ws + off); off += (size_t)DI2 * DM      * 2;  // in_w f16
  _Float16* Wxh  = (_Float16*)(ws + off); off += (size_t)36  * DI      * 2;  // xproj_w f16
  _Float16* Woh  = (_Float16*)(ws + off); off += (size_t)DM  * DI      * 2;  // out_w f16

  dim3 blk(256);
  k_transpose_in<<<(MTOT * DM + 255) / 256, blk, 0, stream>>>(x, Xh);

  for (int mb = 0; mb < 2; ++mb) {
    const float* in_w    = p[1 + mb * 9 + 0];
    const float* conv_w  = p[1 + mb * 9 + 1];
    const float* conv_b  = p[1 + mb * 9 + 2];
    const float* xproj_w = p[1 + mb * 9 + 3];
    const float* dt_w    = p[1 + mb * 9 + 4];
    const float* dt_b    = p[1 + mb * 9 + 5];
    const float* A_log   = p[1 + mb * 9 + 6];
    const float* Dpar    = p[1 + mb * 9 + 7];
    const float* out_w   = p[1 + mb * 9 + 8];
    const float* lnw     = p[19 + mb * 2];
    const float* lnb     = p[20 + mb * 2];

    // convert this block's weights to f16 (tiny)
    k_cvt_f16<<<(DI2 * DM + 255) / 256, blk, 0, stream>>>(in_w,    Wih, DI2 * DM);
    k_cvt_f16<<<(36 * DI + 255)  / 256, blk, 0, stream>>>(xproj_w, Wxh, 36 * DI);
    k_cvt_f16<<<(DM * DI + 255)  / 256, blk, 0, stream>>>(out_w,   Woh, DM * DI);

    {   // in_proj: (M x 64) * (64 x 256)
      int ntilesN = DI2 / 16, ntiles = (MTOT / 16) * ntilesN;
      k_gemm_wmma<64><<<(ntiles + 7) / 8, blk, 0, stream>>>(
          Xh, DM, Wih, DI2, XZ, DI2, ntilesN, ntiles);
    }
    k_conv_silu<<<(MTOT * DI + 255) / 256, blk, 0, stream>>>(XZ, conv_w, conv_b, XC, XCh);
    {   // x_proj: (M x 128) * (128 x 36), N padded to 48
      int ntilesN = DBL_LD / 16, ntiles = (MTOT / 16) * ntilesN;
      k_gemm_wmma<128><<<(ntiles + 7) / 8, blk, 0, stream>>>(
          XCh, DI, Wxh, 36, DBLb, DBL_LD, ntilesN, ntiles);
    }
    k_scan<<<BATCH, DI, 0, stream>>>(DBLb, XC, XZ, dt_w, dt_b, A_log, Dpar, YGh);
    {   // out_proj: (M x 128) * (128 x 64)
      int ntilesN = DM / 16, ntiles = (MTOT / 16) * ntilesN;
      k_gemm_wmma<128><<<(ntiles + 7) / 8, blk, 0, stream>>>(
          YGh, DI, Woh, DM, O1, DM, ntilesN, ntiles);
    }
    if (mb == 0)
      k_ln_relu<<<(MTOT + 7) / 8, blk, 0, stream>>>(O1, lnw, lnb, Xh, nullptr, 0);
    else
      k_ln_relu<<<(MTOT + 7) / 8, blk, 0, stream>>>(O1, lnw, lnb, nullptr, (float*)d_out, 1);
  }
}
